// Net_30176440221873
// MI455X (gfx1250) — matrix operand order repaired and checked by reading, not compiled, not run on hardware
//
#include <hip/hip_runtime.h>
#include <hip/hip_bf16.h>

#define N_NODES 50000
#define N_EDGES 300000
#define ANNOT   200
#define APAD    224        // ANNOT padded to multiple of 32
#define HID     256
#define N_STEPS 8

typedef __attribute__((ext_vector_type(16))) __bf16 v16bf;
typedef __attribute__((ext_vector_type(8)))  float  v8f;

union ABv { uint4 u[2]; v16bf v; };

__device__ __forceinline__ unsigned short f2bf(float f) {
  union { float f; unsigned int u; } c; c.f = f;
  unsigned int u = c.u;
  u += 0x7FFFu + ((u >> 16) & 1u);   // round-to-nearest-even
  return (unsigned short)(u >> 16);
}

// ---------------------------------------------------------------------------
// Pack an fp32 weight matrix into the bf16 B-operand tile layout:
// out[((jt*ktiles + kt)*32 + lane)*16 + i] = B[kt*32 + (lane>>4)*16 + i][jt*16 + (lane&15)]
// where B[k][j] = trans ? W[j*ldw + k] : W[k*ldw + j]; k >= Kreal -> 0.
// ---------------------------------------------------------------------------
__global__ void k_pack_b(const float* __restrict__ W, unsigned short* __restrict__ out,
                         int ktiles, int Jtiles, int Kreal, int ldw, int trans) {
  int tid = blockIdx.x * blockDim.x + threadIdx.x;
  int total = ktiles * Jtiles * 512;
  if (tid >= total) return;
  int i    = tid & 15;
  int lane = (tid >> 4) & 31;
  int rest = tid >> 9;
  int kt = rest % ktiles;
  int jt = rest / ktiles;
  int j = jt * 16 + (lane & 15);
  int k = kt * 32 + (lane >> 4) * 16 + i;
  float v = 0.0f;
  if (k < Kreal) v = trans ? W[(size_t)j * ldw + k] : W[(size_t)k * ldw + j];
  out[tid] = f2bf(v);
}

// fp32 [rows,Kin] -> bf16 [rows,Kpad] with zero padding
__global__ void k_cvt_pad(const float* __restrict__ in, unsigned short* __restrict__ out,
                          int rows, int Kin, int Kpad) {
  int tid = blockIdx.x * blockDim.x + threadIdx.x;
  if (tid >= rows * Kpad) return;
  int col = tid % Kpad;
  int row = tid / Kpad;
  float v = (col < Kin) ? in[(size_t)row * Kin + col] : 0.0f;
  out[tid] = f2bf(v);
}

__global__ void k_zero4(float* __restrict__ p, int n4) {
  int tid = blockIdx.x * blockDim.x + threadIdx.x;
  if (tid < n4) ((float4*)p)[tid] = make_float4(0.f, 0.f, 0.f, 0.f);
}

// agg[dst] += m[src], 4 floats per thread
__global__ void k_scatter(const float* __restrict__ m, const int* __restrict__ ei,
                          float* __restrict__ agg) {
  int tid = blockIdx.x * blockDim.x + threadIdx.x;
  int e = tid >> 6;
  if (e >= N_EDGES) return;
  int c = (tid & 63) << 2;
  int s = ei[e];
  int d = ei[N_EDGES + e];
  float4 v = *(const float4*)(m + (size_t)s * HID + c);
  float* a = agg + (size_t)d * HID + c;
  atomicAdd(a + 0, v.x);
  atomicAdd(a + 1, v.y);
  atomicAdd(a + 2, v.z);
  atomicAdd(a + 3, v.w);
}

// ---------------------------------------------------------------------------
// bf16 WMMA GEMM, B staged through LDS and shared by the 8 waves of a block.
// Block = 8 waves, wave w -> M-tile (blockM*8 + w), all waves same j-group
// (64 columns). B chunk for the j-group (4 jtiles x ktiles x 1KB) is loaded
// into LDS once, then the K-loop reads it with ds_load_b128.
// ---------------------------------------------------------------------------
__global__ void k_gemm(const unsigned short* __restrict__ A,
                       const unsigned short* __restrict__ B,
                       const float* __restrict__ bias,
                       float* __restrict__ C,
                       int Mtiles, int Jgroups, int ktiles, int lda, int ldc) {
  __shared__ uint4 sB[2048];                 // 4 jtiles * (<=8) ktiles * 512 bf16 = 32KB
  int wave = threadIdx.x >> 5;
  int lane = threadIdx.x & 31;
  int jg = blockIdx.x % Jgroups;
  int mb = blockIdx.x / Jgroups;
  int mt = mb * 8 + wave;
  bool active = mt < Mtiles;
  int m0 = mt * 16, j0 = jg * 64;
  int half = lane >> 4, n = lane & 15;

  // cooperative stage of this j-group's B chunk (contiguous in the pack)
  int nu4 = 4 * ktiles * 64;
  const uint4* Bsrc = (const uint4*)B + (size_t)jg * 4 * ktiles * 64;
  for (int q = threadIdx.x; q < nu4; q += blockDim.x) sB[q] = Bsrc[q];
  __syncthreads();

  if (active) {
    v8f acc[4];
#pragma unroll
    for (int t = 0; t < 4; ++t) {
      float bv = bias ? bias[j0 + t * 16 + n] : 0.0f;
#pragma unroll
      for (int r = 0; r < 8; ++r) acc[t][r] = bv;
    }

    const unsigned short* arow = A + (size_t)(m0 + n) * lda + half * 16;
    for (int kt = 0; kt < ktiles; ++kt) {
      ABv a;
      a.u[0] = *(const uint4*)(arow + kt * 32);        // K = kb .. kb+7
      a.u[1] = *(const uint4*)(arow + kt * 32 + 8);    // K = kb+8 .. kb+15
      ABv b[4];
#pragma unroll
      for (int t = 0; t < 4; ++t) {
        const uint4* p = sB + (t * ktiles + kt) * 64 + lane * 2;
        b[t].u[0] = p[0];
        b[t].u[1] = p[1];
      }
#pragma unroll
      for (int t = 0; t < 4; ++t)
        acc[t] = __builtin_amdgcn_wmma_f32_16x16x32_bf16(false, a.v, false, b[t].v,
                                                         (short)0, acc[t], false, false);
    }
#pragma unroll
    for (int t = 0; t < 4; ++t) {
      float* crow = C + (size_t)(m0 + half * 8) * ldc + j0 + t * 16 + n;
#pragma unroll
      for (int r = 0; r < 8; ++r) crow[(size_t)r * ldc] = acc[t][r];
    }
  }
}

// ---------------------------------------------------------------------------
// Fused GRU cell with LDS-staged gate weights. Block = 8 waves, wave w ->
// M-tile (blockM*8 + w). Per 16-column chunk jc, the six B tile-rows
// (i_r,i_z,i_n / h_r,h_z,h_n; 48KB) are staged in LDS once per block, then
// each wave runs the K-loop (6 WMMA per k-step), gates, in-place h update.
// ---------------------------------------------------------------------------
__global__ void k_gru(const unsigned short* __restrict__ aggb,
                      const unsigned short* __restrict__ hb,
                      const unsigned short* __restrict__ Bih,
                      const unsigned short* __restrict__ Bhh,
                      const float* __restrict__ b_ih,
                      const float* __restrict__ b_hh,
                      float* __restrict__ h,
                      int Mtiles) {
  __shared__ uint4 sB[3072];                 // 6 tiles * 8 ktiles * 512 bf16 = 48KB
  int wave = threadIdx.x >> 5;
  int lane = threadIdx.x & 31;
  int mt = blockIdx.x * 8 + wave;
  bool active = mt < Mtiles;
  int m0 = mt * 16;
  int half = lane >> 4, n = lane & 15;
  const int ktiles = HID / 32;               // 8

  const unsigned short* ar = aggb + (size_t)(m0 + n) * HID + half * 16;
  const unsigned short* hr = hb   + (size_t)(m0 + n) * HID + half * 16;

  for (int jc = 0; jc < HID / 16; ++jc) {
    // stage the 6 gate-weight tile-rows for this column chunk (each 512 uint4)
    for (int q = threadIdx.x; q < 3072; q += blockDim.x) {
      int t = q >> 9, u = q & 511;
      const uint4* Bp = (const uint4*)((t < 3) ? Bih : Bhh);
      int jt = (t % 3) * 16 + jc;
      sB[q] = Bp[(size_t)jt * 512 + u];
    }
    __syncthreads();

    if (active) {
      v8f a_r, a_z, a_n, g_r, g_z, g_n;
      {
        int j = jc * 16 + n;
        float bir = b_ih[j], biz = b_ih[HID + j], bin = b_ih[2 * HID + j];
        float bhr = b_hh[j], bhz = b_hh[HID + j], bhn = b_hh[2 * HID + j];
#pragma unroll
        for (int r = 0; r < 8; ++r) {
          a_r[r] = bir; a_z[r] = biz; a_n[r] = bin;
          g_r[r] = bhr; g_z[r] = bhz; g_n[r] = bhn;
        }
      }
      for (int kt = 0; kt < ktiles; ++kt) {
        ABv aa, ah;
        aa.u[0] = *(const uint4*)(ar + kt * 32);
        aa.u[1] = *(const uint4*)(ar + kt * 32 + 8);
        ah.u[0] = *(const uint4*)(hr + kt * 32);
        ah.u[1] = *(const uint4*)(hr + kt * 32 + 8);
        ABv b[6];
#pragma unroll
        for (int t = 0; t < 6; ++t) {
          const uint4* p = sB + t * 512 + kt * 64 + lane * 2;
          b[t].u[0] = p[0];
          b[t].u[1] = p[1];
        }
        a_r = __builtin_amdgcn_wmma_f32_16x16x32_bf16(false, aa.v, false, b[0].v, (short)0, a_r, false, false);
        a_z = __builtin_amdgcn_wmma_f32_16x16x32_bf16(false, aa.v, false, b[1].v, (short)0, a_z, false, false);
        a_n = __builtin_amdgcn_wmma_f32_16x16x32_bf16(false, aa.v, false, b[2].v, (short)0, a_n, false, false);
        g_r = __builtin_amdgcn_wmma_f32_16x16x32_bf16(false, ah.v, false, b[3].v, (short)0, g_r, false, false);
        g_z = __builtin_amdgcn_wmma_f32_16x16x32_bf16(false, ah.v, false, b[4].v, (short)0, g_z, false, false);
        g_n = __builtin_amdgcn_wmma_f32_16x16x32_bf16(false, ah.v, false, b[5].v, (short)0, g_n, false, false);
      }
      float* hrow = h + (size_t)(m0 + half * 8) * HID + jc * 16 + n;
#pragma unroll
      for (int r = 0; r < 8; ++r) {
        float rg = 1.0f / (1.0f + __expf(-(a_r[r] + g_r[r])));
        float zg = 1.0f / (1.0f + __expf(-(a_z[r] + g_z[r])));
        float ng = tanhf(a_n[r] + rg * g_n[r]);
        float hv = hrow[(size_t)r * HID];
        hrow[(size_t)r * HID] = (1.0f - zg) * ng + zg * hv;
      }
    }
    __syncthreads();   // protect LDS rewrite on next jc
  }
}

// out = log_softmax(relu(h) @ W_lin^T + b_lin)  ([N,2])
__global__ void k_out(const float* __restrict__ h, const float* __restrict__ W_lin,
                      const float* __restrict__ b_lin, float* __restrict__ out) {
  int nid = blockIdx.x * blockDim.x + threadIdx.x;
  if (nid >= N_NODES) return;
  const float4* hp = (const float4*)(h + (size_t)nid * HID);
  const float4* w0 = (const float4*)(W_lin);
  const float4* w1 = (const float4*)(W_lin + HID);
  float l0 = 0.f, l1 = 0.f;
  for (int k = 0; k < HID / 4; ++k) {
    float4 hv = hp[k];
    hv.x = fmaxf(hv.x, 0.f); hv.y = fmaxf(hv.y, 0.f);
    hv.z = fmaxf(hv.z, 0.f); hv.w = fmaxf(hv.w, 0.f);
    float4 a = w0[k], b = w1[k];
    l0 += hv.x * a.x + hv.y * a.y + hv.z * a.z + hv.w * a.w;
    l1 += hv.x * b.x + hv.y * b.y + hv.z * b.z + hv.w * b.w;
  }
  l0 += b_lin[0];
  l1 += b_lin[1];
  float mx  = fmaxf(l0, l1);
  float lse = mx + __logf(__expf(l0 - mx) + __expf(l1 - mx));
  out[nid * 2 + 0] = l0 - lse;
  out[nid * 2 + 1] = l1 - lse;
}

extern "C" void kernel_launch(void* const* d_in, const int* in_sizes, int n_in,
                              void* d_out, int out_size, void* d_ws, size_t ws_size,
                              hipStream_t stream) {
  const float* x        = (const float*)d_in[0];
  const int*   ei       = (const int*)d_in[1];
  // d_in[2] = batch (unused)
  const float* W_reduce = (const float*)d_in[3];
  const float* b_reduce = (const float*)d_in[4];
  const float* W_ggc    = (const float*)d_in[5];
  const float* W_ih     = (const float*)d_in[6];
  const float* W_hh     = (const float*)d_in[7];
  const float* b_ih     = (const float*)d_in[8];
  const float* b_hh     = (const float*)d_in[9];
  const float* W_lin    = (const float*)d_in[10];
  const float* b_lin    = (const float*)d_in[11];
  float* out = (float*)d_out;

  char* ws = (char*)d_ws;
  size_t off = 0;
  auto take = [&](size_t bytes) -> char* {
    char* p = ws + off;
    off = (off + bytes + 255) & ~(size_t)255;
    return p;
  };
  float*          h    = (float*)take((size_t)N_NODES * HID * 4);
  float*          m    = (float*)take((size_t)N_NODES * HID * 4);
  float*          agg  = (float*)take((size_t)N_NODES * HID * 4);
  unsigned short* hb   = (unsigned short*)take((size_t)N_NODES * HID * 2);
  unsigned short* aggb = (unsigned short*)take((size_t)N_NODES * HID * 2);
  unsigned short* xb   = (unsigned short*)take((size_t)N_NODES * APAD * 2);
  unsigned short* Bred = (unsigned short*)take((size_t)APAD * HID * 2);
  unsigned short* Bm   = (unsigned short*)take((size_t)N_STEPS * HID * HID * 2);
  unsigned short* Bih  = (unsigned short*)take((size_t)HID * 3 * HID * 2);
  unsigned short* Bhh  = (unsigned short*)take((size_t)HID * 3 * HID * 2);

  const int TB = 256;
  const int Mtiles = N_NODES / 16;           // 3125 (exact)
  const int Mblocks = (Mtiles + 7) / 8;      // 391

  // --- weight packing (once per launch, deterministic) ---
  k_pack_b<<<(7 * 16 * 512 + TB - 1) / TB, TB, 0, stream>>>(W_reduce, Bred, 7, 16, ANNOT, ANNOT, 1);
  for (int s = 0; s < N_STEPS; ++s)
    k_pack_b<<<(8 * 16 * 512 + TB - 1) / TB, TB, 0, stream>>>(W_ggc + (size_t)s * HID * HID,
                                                              Bm + (size_t)s * HID * HID,
                                                              8, 16, HID, HID, 0);
  k_pack_b<<<(8 * 48 * 512 + TB - 1) / TB, TB, 0, stream>>>(W_ih, Bih, 8, 48, HID, HID, 1);
  k_pack_b<<<(8 * 48 * 512 + TB - 1) / TB, TB, 0, stream>>>(W_hh, Bhh, 8, 48, HID, HID, 1);

  // --- h = x @ W_reduce^T + b_reduce ---
  k_cvt_pad<<<((size_t)N_NODES * APAD + TB - 1) / TB, TB, 0, stream>>>(x, xb, N_NODES, ANNOT, APAD);
  k_gemm<<<Mblocks * 4, TB, 0, stream>>>(xb, Bred, b_reduce, h, Mtiles, 4, 7, APAD, HID);

  // --- 8 GGNN steps ---
  for (int s = 0; s < N_STEPS; ++s) {
    k_cvt_pad<<<((size_t)N_NODES * HID + TB - 1) / TB, TB, 0, stream>>>(h, hb, N_NODES, HID, HID);
    k_gemm<<<Mblocks * 4, TB, 0, stream>>>(hb, Bm + (size_t)s * HID * HID, nullptr, m,
                                           Mtiles, 4, 8, HID, HID);
    k_zero4<<<((size_t)N_NODES * HID / 4 + TB - 1) / TB, TB, 0, stream>>>(agg, N_NODES * HID / 4);
    k_scatter<<<((size_t)N_EDGES * 64 + TB - 1) / TB, TB, 0, stream>>>(m, ei, agg);
    k_cvt_pad<<<((size_t)N_NODES * HID + TB - 1) / TB, TB, 0, stream>>>(agg, aggb, N_NODES, HID, HID);
    k_gru<<<Mblocks, TB, 0, stream>>>(aggb, hb, Bih, Bhh, b_ih, b_hh, h, Mtiles);
  }

  // --- head ---
  k_out<<<(N_NODES + TB - 1) / TB, TB, 0, stream>>>(h, W_lin, b_lin, out);
}
